// RMHA_48747878810082
// MI455X (gfx1250) — matrix operand order repaired and checked by reading, not compiled, not run on hardware
//
#include <hip/hip_runtime.h>
#include <math.h>

// ---------------- types ----------------
typedef __attribute__((ext_vector_type(16))) __bf16 v16bf;
typedef __attribute__((ext_vector_type(8)))  __bf16 bf16x8;
typedef __attribute__((ext_vector_type(8)))  float  v8f;

#define MODE_F32 0
#define MODE_P   1
#define MODE_QKV 2
#define MODE_E   3

static __device__ __forceinline__ v8f vzero8() {
    v8f v;
#pragma unroll
    for (int i = 0; i < 8; ++i) v[i] = 0.0f;
    return v;
}

static __device__ __forceinline__ v16bf make_frag(const __bf16* lo, const __bf16* hi) {
    v16bf r;
    *reinterpret_cast<bf16x8*>(&r)       = *reinterpret_cast<const bf16x8*>(lo);
    *(reinterpret_cast<bf16x8*>(&r) + 1) = *reinterpret_cast<const bf16x8*>(hi);
    return r;
}

static __device__ __forceinline__ v8f wmma_bf16(v16bf a, v16bf b, v8f c) {
    return __builtin_amdgcn_wmma_f32_16x16x32_bf16(false, a, false, b, (short)0, c, false, false);
}

// CDNA5 async DMA: global -> LDS, 16 bytes per lane, tracked by ASYNCcnt.
// LDS operand is the low 32 bits of the generic pointer.
static __device__ __forceinline__ void async_copy16(const __bf16* gsrc, __bf16* ldst) {
    unsigned lds_addr = (unsigned)(uintptr_t)ldst;
    unsigned long long gaddr = (unsigned long long)(uintptr_t)gsrc;
    asm volatile("global_load_async_to_lds_b128 %0, %1, off"
                 :: "v"(lds_addr), "v"(gaddr) : "memory");
}

static __device__ __forceinline__ void wait_async0() {
    asm volatile("s_wait_asynccnt 0" ::: "memory");
}

// ---------------- LayerNorm -> bf16 ----------------
__global__ __launch_bounds__(256)
void ln_kernel(const float* __restrict__ x, const float* __restrict__ g,
               const float* __restrict__ be, __bf16* __restrict__ xn) {
    int row = blockIdx.x, tid = threadIdx.x;
    const float* xr = x + (size_t)row * 512;
    float a0 = xr[tid], a1 = xr[tid + 256];
    __shared__ float s1[256], s2[256];
    s1[tid] = a0 + a1; s2[tid] = a0 * a0 + a1 * a1;
    __syncthreads();
    for (int o = 128; o > 0; o >>= 1) {
        if (tid < o) { s1[tid] += s1[tid + o]; s2[tid] += s2[tid + o]; }
        __syncthreads();
    }
    float mean = s1[0] * (1.0f / 512.0f);
    float var  = s2[0] * (1.0f / 512.0f) - mean * mean;
    float inv  = rsqrtf(var + 1e-5f);
    xn[(size_t)row * 512 + tid]       = (__bf16)((a0 - mean) * inv * g[tid] + be[tid]);
    xn[(size_t)row * 512 + tid + 256] = (__bf16)((a1 - mean) * inv * g[tid + 256] + be[tid + 256]);
}

// ---------------- sinusoidal PE -> bf16 ----------------
__global__ __launch_bounds__(256)
void pe_kernel(__bf16* __restrict__ pe) {
    int t = blockIdx.x, i = threadIdx.x;             // 256 sin/cos pairs
    float freq = __expf(-logf(10000.0f) * (2.0f * (float)i) / 512.0f);
    float ang = (float)t * freq;
    float s, c;
    __sincosf(ang, &s, &c);
    pe[(size_t)t * 512 + 2 * i]     = (__bf16)s;
    pe[(size_t)t * 512 + 2 * i + 1] = (__bf16)c;
}

// ---------------- weight convert + transpose: w[K][N] f32 -> wt[N][K] bf16 ----------------
__global__ __launch_bounds__(256)
void conv_t_kernel(const float* __restrict__ w, __bf16* __restrict__ wt, int K, int N) {
    int idx = blockIdx.x * 256 + threadIdx.x;
    if (idx >= K * N) return;
    int k = idx / N, n = idx % N;
    wt[(size_t)n * K + k] = (__bf16)w[idx];
}

// ---------------- generic 128x128 WMMA bf16 GEMM: C = A[M,K] * Bt[N,K]^T ----------------
__global__ __launch_bounds__(256)
void gemm_kernel(const __bf16* __restrict__ A, int lda, long strideA,
                 const __bf16* __restrict__ Bt, int ldb, int strideBh, long strideBz,
                 const float* __restrict__ bias,
                 void* out0, void* out1, void* out2,
                 int M, int N, int K, int mode) {
    __shared__ __bf16 Asm[128][40];
    __shared__ __bf16 Bsm[128][40];
    int z = blockIdx.z;
    const __bf16* Ag = A + (size_t)z * strideA;
    const __bf16* Bg = Bt + (size_t)z * strideBz + (size_t)(z & 7) * strideBh;
    int row0 = blockIdx.y * 128, col0 = blockIdx.x * 128;
    int tid = threadIdx.x, lane = tid & 31, w = tid >> 5;
    int wm = w >> 2, wn = w & 3;
    int m16 = lane & 15, hh = lane >> 4;

    // each thread owns two 8-element chunks of the 128x32 tiles
    int c0 = tid, c1 = tid + 256;
    int r0 = c0 >> 2, q0c = c0 & 3;
    int r1 = c1 >> 2, q1c = c1 & 3;

    v8f acc[4][2];
#pragma unroll
    for (int mi = 0; mi < 4; ++mi)
#pragma unroll
        for (int ni = 0; ni < 2; ++ni) acc[mi][ni] = vzero8();

    for (int kk = 0; kk < K; kk += 32) {
        // CDNA5 async DMA global->LDS (no VGPR round-trip)
        async_copy16(&Ag[(size_t)(row0 + r0) * lda + kk + q0c * 8], &Asm[r0][q0c * 8]);
        async_copy16(&Bg[(size_t)(col0 + r0) * ldb + kk + q0c * 8], &Bsm[r0][q0c * 8]);
        async_copy16(&Ag[(size_t)(row0 + r1) * lda + kk + q1c * 8], &Asm[r1][q1c * 8]);
        async_copy16(&Bg[(size_t)(col0 + r1) * ldb + kk + q1c * 8], &Bsm[r1][q1c * 8]);
        wait_async0();
        __syncthreads();

        v16bf af[4], bfr[2];
#pragma unroll
        for (int mi = 0; mi < 4; ++mi) {
            const __bf16* rp = &Asm[wm * 64 + mi * 16 + m16][0];
            af[mi] = make_frag(rp + 8 * hh, rp + 16 + 8 * hh);
        }
#pragma unroll
        for (int ni = 0; ni < 2; ++ni) {
            const __bf16* rp = &Bsm[wn * 32 + ni * 16 + m16][0];
            bfr[ni] = make_frag(rp + 8 * hh, rp + 16 + 8 * hh);
        }
#pragma unroll
        for (int mi = 0; mi < 4; ++mi)
#pragma unroll
            for (int ni = 0; ni < 2; ++ni)
                acc[mi][ni] = wmma_bf16(af[mi], bfr[ni], acc[mi][ni]);
        __syncthreads();
    }

#pragma unroll
    for (int mi = 0; mi < 4; ++mi)
#pragma unroll
        for (int ni = 0; ni < 2; ++ni)
#pragma unroll
            for (int r = 0; r < 8; ++r) {
                int row = row0 + wm * 64 + mi * 16 + 8 * hh + r;
                int col = col0 + wn * 32 + ni * 16 + m16;
                float v = acc[mi][ni][r];
                if (mode == MODE_F32) {
                    ((float*)out0)[(size_t)row * N + col] = v + bias[col];
                } else if (mode == MODE_P) {
                    ((__bf16*)out0)[(size_t)row * N + col] = (__bf16)v;
                } else if (mode == MODE_E) {
                    ((__bf16*)out0)[(size_t)z * 2048 * 2048 + (size_t)row * 2048 + col] = (__bf16)v;
                } else { // MODE_QKV: scatter into Q[z][t][64], K[z][t][64], Vt[z][64][t]
                    v += bias[col];
                    int part = col >> 9, rem = col & 511;
                    int h = rem >> 6, d = rem & 63;
                    int b = row >> 11, t = row & 2047;
                    size_t zz = (size_t)(b * 8 + h);
                    if (part == 0)       ((__bf16*)out0)[(zz * 2048 + t) * 64 + d] = (__bf16)v;
                    else if (part == 1)  ((__bf16*)out1)[(zz * 2048 + t) * 64 + d] = (__bf16)v;
                    else                 ((__bf16*)out2)[(zz * 64 + d) * 2048 + t] = (__bf16)v;
                }
            }
}

// ---------------- fused flash attention with Transformer-XL skew gather ----------------
__global__ __launch_bounds__(256)
void attn_kernel(const __bf16* __restrict__ Q, const __bf16* __restrict__ Km,
                 const __bf16* __restrict__ Vt, const __bf16* __restrict__ E,
                 __bf16* __restrict__ O) {
    __shared__ __bf16 P_lds[8][16][136];
    const int T = 2048;
    int z = blockIdx.y;                                  // b*8 + head
    int tid = threadIdx.x, lane = tid & 31, w = tid >> 5;
    int q0 = blockIdx.x * 128 + w * 16;                  // 16 query rows per wave
    int n = lane & 15, hh = lane >> 4;
    const float inv_s = 0.044194173824159216f;           // 1/sqrt(512)

    // Q fragments (A-matrix: lane%16 = row, halves select k-chunks)
    v16bf qf[2];
    {
        const __bf16* qrow = Q + ((size_t)z * T + q0 + n) * 64;
        qf[0] = make_frag(qrow + 8 * hh,       qrow + 16 + 8 * hh);
        qf[1] = make_frag(qrow + 32 + 8 * hh,  qrow + 48 + 8 * hh);
    }

    float mrun[8], lrun[8];
    v8f oacc[4];
#pragma unroll
    for (int r = 0; r < 8; ++r) { mrun[r] = -1e30f; lrun[r] = 0.0f; }
#pragma unroll
    for (int d = 0; d < 4; ++d) oacc[d] = vzero8();

    const size_t ebase = (size_t)z * T * T;

    for (int j0 = 0; j0 < T; j0 += 128) {
        // prefetch next k-tile (emits global_prefetch_b8)
        if (j0 + 128 < T) {
            __builtin_prefetch(Km + ((size_t)z * T + j0 + 128 + lane * 4) * 64, 0, 0);
            __builtin_prefetch(Vt + ((size_t)z * 64 + lane * 2) * T + j0 + 128, 0, 0);
        }
        float sv[8][8];
#pragma unroll
        for (int s = 0; s < 8; ++s) {
            v8f a = vzero8();
            int kb = j0 + s * 16;
            const __bf16* krow = Km + ((size_t)z * T + kb + n) * 64;
            v16bf b0 = make_frag(krow + 8 * hh,      krow + 16 + 8 * hh);
            v16bf b1 = make_frag(krow + 32 + 8 * hh, krow + 48 + 8 * hh);
            a = wmma_bf16(qf[0], b0, a);
            a = wmma_bf16(qf[1], b1, a);
            int k = kb + n;
#pragma unroll
            for (int r = 0; r < 8; ++r) {
                int q = q0 + 8 * hh + r;
                float e;
                if (k <= q)          e = (float)E[ebase + (size_t)q * T + (T - 1 + k - q)];
                else if (k == q + 1) e = 0.0f;
                else                 e = (float)E[ebase + (size_t)(q + 1) * T + (k - q - 2)];
                sv[s][r] = (a[r] + e) * inv_s;
            }
        }

        // online softmax per row (rows live in lane-halves; reduce across 16 lanes)
        float scl[8];
#pragma unroll
        for (int r = 0; r < 8; ++r) {
            float tm = sv[0][r];
#pragma unroll
            for (int s = 1; s < 8; ++s) tm = fmaxf(tm, sv[s][r]);
#pragma unroll
            for (int msk = 1; msk < 16; msk <<= 1) tm = fmaxf(tm, __shfl_xor(tm, msk, 32));
            float mnew = fmaxf(mrun[r], tm);
            scl[r] = __expf(mrun[r] - mnew);
            mrun[r] = mnew;
            float rs = 0.0f;
#pragma unroll
            for (int s = 0; s < 8; ++s) { float p = __expf(sv[s][r] - mnew); sv[s][r] = p; rs += p; }
#pragma unroll
            for (int msk = 1; msk < 16; msk <<= 1) rs += __shfl_xor(rs, msk, 32);
            lrun[r] = lrun[r] * scl[r] + rs;
        }

        // stage probs (C-layout) into per-wave LDS, re-read in A-fragment layout
#pragma unroll
        for (int s = 0; s < 8; ++s)
#pragma unroll
            for (int r = 0; r < 8; ++r)
                P_lds[w][8 * hh + r][s * 16 + n] = (__bf16)sv[s][r];

#pragma unroll
        for (int d = 0; d < 4; ++d)
#pragma unroll
            for (int r = 0; r < 8; ++r) oacc[d][r] *= scl[r];

        asm volatile("s_wait_dscnt 0" ::: "memory");

        v16bf pf[4];
#pragma unroll
        for (int kc = 0; kc < 4; ++kc) {
            const __bf16* prow = &P_lds[w][n][kc * 32];
            pf[kc] = make_frag(prow + 8 * hh, prow + 16 + 8 * hh);
        }
#pragma unroll
        for (int ds = 0; ds < 4; ++ds)
#pragma unroll
            for (int kc = 0; kc < 4; ++kc) {
                const __bf16* vrow = Vt + ((size_t)z * 64 + ds * 16 + n) * T + j0 + kc * 32;
                v16bf vf = make_frag(vrow + 8 * hh, vrow + 16 + 8 * hh);
                oacc[ds] = wmma_bf16(pf[kc], vf, oacc[ds]);
            }
    }

    int b = z >> 3, head = z & 7;
#pragma unroll
    for (int ds = 0; ds < 4; ++ds)
#pragma unroll
        for (int r = 0; r < 8; ++r) {
            int q = q0 + 8 * hh + r;
            int d = ds * 16 + n;
            O[((size_t)b * T + q) * 512 + head * 64 + d] = (__bf16)(oacc[ds][r] / lrun[r]);
        }
}

// ---------------- host ----------------
extern "C" void kernel_launch(void* const* d_in, const int* in_sizes, int n_in,
                              void* d_out, int out_size, void* d_ws, size_t ws_size,
                              hipStream_t stream) {
    (void)in_sizes; (void)n_in; (void)out_size; (void)ws_size;
    const float* x      = (const float*)d_in[0];
    const float* w_qkv  = (const float*)d_in[1];
    const float* b_qkv  = (const float*)d_in[2];
    const float* w_pos  = (const float*)d_in[3];
    const float* w_ffn  = (const float*)d_in[4];
    const float* b_ffn  = (const float*)d_in[5];
    const float* ln_g   = (const float*)d_in[6];
    const float* ln_b   = (const float*)d_in[7];
    float* out = (float*)d_out;

    char* ws = (char*)d_ws;
    size_t off = 0;
    auto take = [&](size_t bytes) -> char* {
        char* p = ws + off;
        off = (off + bytes + 255) & ~(size_t)255;
        return p;
    };
    __bf16* xn  = (__bf16*)take(4096ull * 512 * 2);
    __bf16* peb = (__bf16*)take(2048ull * 512 * 2);
    __bf16* wqt = (__bf16*)take(1536ull * 512 * 2);
    __bf16* wpt = (__bf16*)take(512ull * 512 * 2);
    __bf16* wft = (__bf16*)take(512ull * 512 * 2);
    __bf16* Qm  = (__bf16*)take(16ull * 2048 * 64 * 2);
    __bf16* Km  = (__bf16*)take(16ull * 2048 * 64 * 2);
    __bf16* Vt  = (__bf16*)take(16ull * 2048 * 64 * 2);
    __bf16* Pm  = (__bf16*)take(2048ull * 512 * 2);
    __bf16* Om  = (__bf16*)take(4096ull * 512 * 2);
    __bf16* Em  = (__bf16*)take(16ull * 2048 * 2048 * 2);   // ~134 MB, mostly L2-resident

    ln_kernel<<<4096, 256, 0, stream>>>(x, ln_g, ln_b, xn);
    pe_kernel<<<2048, 256, 0, stream>>>(peb);
    conv_t_kernel<<<(512 * 1536 + 255) / 256, 256, 0, stream>>>(w_qkv, wqt, 512, 1536);
    conv_t_kernel<<<(512 * 512 + 255) / 256, 256, 0, stream>>>(w_pos, wpt, 512, 512);
    conv_t_kernel<<<(512 * 512 + 255) / 256, 256, 0, stream>>>(w_ffn, wft, 512, 512);

    // QKV: xn[4096,512] @ w_qkv -> Q/K/Vt scatter (+bias)
    gemm_kernel<<<dim3(12, 32, 1), 256, 0, stream>>>(xn, 512, 0, wqt, 512, 0, 0,
                                                     b_qkv, Qm, Km, Vt, 4096, 1536, 512, MODE_QKV);
    // P: pe[2048,512] @ w_pos -> bf16 [2048,512]
    gemm_kernel<<<dim3(4, 16, 1), 256, 0, stream>>>(peb, 512, 0, wpt, 512, 0, 0,
                                                    nullptr, Pm, nullptr, nullptr, 2048, 512, 512, MODE_P);
    // E[z] = Q[z] @ P(head z%8)^T, batched over 16 heads
    gemm_kernel<<<dim3(16, 16, 16), 256, 0, stream>>>(Qm, 64, 2048l * 64, Pm, 512, 64, 0,
                                                      nullptr, Em, nullptr, nullptr, 2048, 2048, 64, MODE_E);
    // fused flash attention with skew gather
    attn_kernel<<<dim3(16, 16, 1), 256, 0, stream>>>(Qm, Km, Vt, Em, Om);
    // FFN: O[4096,512] @ w_ffn + b_ffn -> d_out (f32)
    gemm_kernel<<<dim3(4, 32, 1), 256, 0, stream>>>(Om, 512, 0, wft, 512, 0, 0,
                                                    b_ffn, out, nullptr, nullptr, 4096, 512, 512, MODE_F32);
}
